// SWA_33225867002140
// MI455X (gfx1250) — matrix operand order repaired and checked by reading, not compiled, not run on hardware
//
#include <hip/hip_runtime.h>
#include <hip/hip_bf16.h>

// ---------------------------------------------------------------------------
// Sliding-window causal attention, fused flash-style, bf16 WMMA on gfx1250.
//   B=4, T=4096, E=512, D=64, WINDOW=256. wave32 only.
//
// Pipeline:
//   K0a: x fp32 -> bf16 (one pass; kills per-tile convert storms)
//   K0b: Wk/Wq/Wv fp32 [E,D] -> bf16 W^T [D,E]  (B-operand = contiguous 32B load)
//   K1 : QKV projection, v_wmma_f32_16x16x32_bf16, V stored transposed [B,D,T]
//   K2 : fused flash attention, 8 WMMA per 32-key tile, online softmax
// ---------------------------------------------------------------------------

typedef __attribute__((ext_vector_type(16))) __bf16 v16bf;
typedef __attribute__((ext_vector_type(8)))  __bf16 v8bf;
typedef __attribute__((ext_vector_type(8)))  float  v8f;

#define T_SEQ   4096
#define E_DIM   512
#define D_HEAD  64
#define WIN     256
#define NEGBIG  (-1e30f)

static __device__ inline __bf16 f2bf(float f) {
    unsigned u = __builtin_bit_cast(unsigned, f);
    unsigned r = u + 0x7FFFu + ((u >> 16) & 1u);     // round-to-nearest-even
    unsigned short h = (unsigned short)(r >> 16);
    return __builtin_bit_cast(__bf16, h);
}

// ---------------------------------------------------------------------------
// Kernel 0a: convert x [B*T*E] fp32 -> bf16, 8 elements/thread, fully coalesced.
// ---------------------------------------------------------------------------
__global__ __launch_bounds__(128)
void swa_convert_x(const float* __restrict__ x, __bf16* __restrict__ xbf)
{
    const size_t i = ((size_t)blockIdx.x * 128 + threadIdx.x) * 8;
    v8f v = *(const v8f*)(x + i);
    v8bf o;
#pragma unroll
    for (int j = 0; j < 8; ++j) o[j] = f2bf(v[j]);
    *(v8bf*)(xbf + i) = o;
}

// ---------------------------------------------------------------------------
// Kernel 0b: W [E=512, D=64] fp32 -> W^T [D=64, E=512] bf16 (coalesced writes).
//   grid = (128, 1, 3); 256 threads.
// ---------------------------------------------------------------------------
__global__ __launch_bounds__(256)
void swa_transpose_w(const float* __restrict__ Wk,
                     const float* __restrict__ Wq,
                     const float* __restrict__ Wv,
                     __bf16* __restrict__ WT)     // [3][64][512]
{
    const int w = blockIdx.z;
    const float* W = (w == 0) ? Wk : ((w == 1) ? Wq : Wv);
    const int o = blockIdx.x * 256 + threadIdx.x;   // 0..32767
    const int e = o & (E_DIM - 1);
    const int d = o >> 9;
    WT[(size_t)w * D_HEAD * E_DIM + (size_t)d * E_DIM + e] = f2bf(W[(size_t)e * D_HEAD + d]);
}

// ---------------------------------------------------------------------------
// Kernel 1: QKV projection  [B*T, E] x [E, D], all-bf16 operand loads.
//   grid = (M/16, 1, 3), block = 128 (4 waves, each a distinct 16-wide N tile).
// ---------------------------------------------------------------------------
__global__ __launch_bounds__(128)
void swa_qkv_proj(const __bf16* __restrict__ xbf,
                  const __bf16* __restrict__ WT,   // [3][64][512]
                  __bf16* __restrict__ kB,         // [B*T, 64] row-major
                  __bf16* __restrict__ qB,         // [B*T, 64] row-major
                  __bf16* __restrict__ vT)         // [B, 64, T] (transposed)
{
    const int lane = threadIdx.x & 31;
    const int half = lane >> 4;            // 0: lanes 0-15, 1: lanes 16-31
    const int l16  = lane & 15;
    const int nt   = threadIdx.x >> 5;     // N tile 0..3
    const int mt   = blockIdx.x;           // M tile 0..1023
    const int w    = blockIdx.z;           // 0:K  1:Q  2:V

    const int row  = mt * 16 + l16;        // A-matrix row for this lane
    const int ncol = nt * 16 + l16;        // B/C column for this lane

    const __bf16* xrow = xbf + (size_t)row * E_DIM;
    const __bf16* wrow = WT + (size_t)w * D_HEAD * E_DIM + (size_t)ncol * E_DIM;

    v8f acc = {};
    for (int kk = 0; kk < E_DIM; kk += 32) {
        // A (16x32): lanes<16 hold K kk..kk+15; lanes>=16 hold K kk+16..kk+31
        const int c0 = kk + half * 16;
        v8bf a0 = *(const v8bf*)(xrow + c0);
        v8bf a1 = *(const v8bf*)(xrow + c0 + 8);
        v16bf a = __builtin_shufflevector(a0, a1, 0,1,2,3,4,5,6,7,8,9,10,11,12,13,14,15);

        // B (32x16): lane holds column ncol, K = kk + half*16 + 0..15 -> one 32B load
        v16bf bm = *(const v16bf*)(wrow + kk + half * 16);

        acc = __builtin_amdgcn_wmma_f32_16x16x32_bf16(false, a, false, bm,
                                                      (short)0, acc, false, false);
    }

    // C layout: VGPR r holds row (r + half*8), column ncol
    if (w < 2) {
        __bf16* dst = (w == 0) ? kB : qB;
#pragma unroll
        for (int r = 0; r < 8; ++r)
            dst[(size_t)(mt * 16 + r + half * 8) * D_HEAD + ncol] = f2bf(acc[r]);
    } else {
#pragma unroll
        for (int r = 0; r < 8; ++r) {
            const int gm = mt * 16 + r + half * 8;        // flat B*T row
            const int bb = gm >> 12;                      // /4096
            const int t  = gm & (T_SEQ - 1);
            vT[((size_t)bb * D_HEAD + ncol) * T_SEQ + t] = f2bf(acc[r]);
        }
    }
}

// ---------------------------------------------------------------------------
// Kernel 2: fused sliding-window flash attention.
//   grid = 256, block = 128: each wave owns one 16-query tile (1024 tiles).
//   Per 32-key tile: 4 WMMA (S = Q K^T), online softmax, P restriped via LDS,
//   4 WMMA (O += P V, full K=32).
// ---------------------------------------------------------------------------
__global__ __launch_bounds__(128)
void swa_attention(const __bf16* __restrict__ qB,
                   const __bf16* __restrict__ kB,
                   const __bf16* __restrict__ vT,
                   float* __restrict__ out)
{
    __shared__ unsigned short pbuf[4 * 512];     // per-wave 16x32 bf16 P tile

    const int lane = threadIdx.x & 31;
    const int half = lane >> 4;
    const int l16  = lane & 15;
    const int wv   = threadIdx.x >> 5;
    const int tile = blockIdx.x * 4 + wv;        // 0..1023
    const int b    = tile >> 8;                  // batch
    const int qbase = (tile & 255) * 16;

    __bf16* P = (__bf16*)(pbuf + wv * 512);

    // ---- load Q tile once (A layout, two 32-wide K chunks over D=64)
    const size_t gq = ((size_t)b * T_SEQ + qbase + l16) * D_HEAD;
    v16bf aq[2];
#pragma unroll
    for (int c = 0; c < 2; ++c) {
        const int c0 = c * 32 + half * 16;
        v8bf q0 = *(const v8bf*)(qB + gq + c0);
        v8bf q1 = *(const v8bf*)(qB + gq + c0 + 8);
        aq[c] = __builtin_shufflevector(q0, q1, 0,1,2,3,4,5,6,7,8,9,10,11,12,13,14,15);
    }

    float mrow[8], lrow[8];
    v8f O[4] = {};
#pragma unroll
    for (int r = 0; r < 8; ++r) { mrow[r] = NEGBIG; lrow[r] = 0.f; }

    const int lo  = qbase - (WIN - 1);                 // lowest key any row may see
    const int kt0 = (lo > 0) ? (lo >> 5) : 0;
    const int kt1 = (qbase + 15) >> 5;

    const __bf16* kbat = kB + (size_t)b * T_SEQ * D_HEAD;
    const __bf16* vbat = vT + (size_t)b * D_HEAD * T_SEQ;

    for (int kt = kt0; kt <= kt1; ++kt) {
        const int kb = kt * 32;

        if (kb + 32 < T_SEQ)   // warm next key tile
            __builtin_prefetch(kbat + (size_t)(kb + 32) * D_HEAD, 0, 1);

        // ---- S = Q K^T  (two 16-key sub-tiles x two 32-wide D chunks)
        v8f S[2] = {};
#pragma unroll
        for (int t2 = 0; t2 < 2; ++t2) {
            const __bf16* krow = kbat + (size_t)(kb + t2 * 16 + l16) * D_HEAD;
#pragma unroll
            for (int c = 0; c < 2; ++c) {
                v16bf bk = *(const v16bf*)(krow + c * 32 + half * 16);
                S[t2] = __builtin_amdgcn_wmma_f32_16x16x32_bf16(false, aq[c], false, bk,
                                                                (short)0, S[t2], false, false);
            }
        }

        // ---- scale + causal/window mask (branch-free; finite sentinel)
#pragma unroll
        for (int r = 0; r < 8; ++r) {
            const int qi = qbase + r + half * 8;
#pragma unroll
            for (int t2 = 0; t2 < 2; ++t2) {
                const int kj = kb + t2 * 16 + l16;
                const bool ok = (kj <= qi) && (kj > qi - WIN);
                S[t2][r] = ok ? (S[t2][r] * 0.125f) : NEGBIG;
            }
        }

        // ---- online softmax per row-register (rows live across 16-lane halves)
#pragma unroll
        for (int r = 0; r < 8; ++r) {
            float mx = fmaxf(S[0][r], S[1][r]);
#pragma unroll
            for (int off = 1; off < 16; off <<= 1)
                mx = fmaxf(mx, __shfl_xor(mx, off, 32));
            const float mnew  = fmaxf(mrow[r], mx);
            const float msafe = (mnew < -1e29f) ? 0.f : mnew;  // all-masked guard
            const float corr  = __expf(mrow[r] - msafe);
            const float p0    = __expf(S[0][r] - msafe);
            const float p1    = __expf(S[1][r] - msafe);
            float sum = p0 + p1;
#pragma unroll
            for (int off = 1; off < 16; off <<= 1)
                sum += __shfl_xor(sum, off, 32);
            lrow[r] = lrow[r] * corr + sum;
            mrow[r] = mnew;
#pragma unroll
            for (int nt = 0; nt < 4; ++nt) O[nt][r] *= corr;

            // stash P (bf16, row-major 16x32) for A-layout reload
            const int rm = r + half * 8;
            P[rm * 32 + l16]      = f2bf(p0);
            P[rm * 32 + 16 + l16] = f2bf(p1);
        }

        asm volatile("s_wait_dscnt 0" ::: "memory");

        // ---- reload P in A layout (16x32, K = 32 keys)
        v16bf aP;
        {
            const int c0 = half * 16;
            v8bf p0 = *(const v8bf*)(P + l16 * 32 + c0);
            v8bf p1 = *(const v8bf*)(P + l16 * 32 + c0 + 8);
            aP = __builtin_shufflevector(p0, p1, 0,1,2,3,4,5,6,7,8,9,10,11,12,13,14,15);
        }

        // ---- O += P V : V^T gives contiguous 32B per-lane B-operand loads
        const __bf16* vkb = vbat + kb + half * 16;
#pragma unroll
        for (int nt = 0; nt < 4; ++nt) {
            const int d = nt * 16 + l16;
            v16bf bv = *(const v16bf*)(vkb + (size_t)d * T_SEQ);
            O[nt] = __builtin_amdgcn_wmma_f32_16x16x32_bf16(false, aP, false, bv,
                                                            (short)0, O[nt], false, false);
        }
    }

    // ---- normalize and store fp32 output [B,T,D]
#pragma unroll
    for (int r = 0; r < 8; ++r) {
        const float inv  = 1.0f / lrow[r];
        const int   rowq = qbase + r + half * 8;
#pragma unroll
        for (int nt = 0; nt < 4; ++nt)
            out[((size_t)b * T_SEQ + rowq) * D_HEAD + nt * 16 + l16] = O[nt][r] * inv;
    }
}

// ---------------------------------------------------------------------------
extern "C" void kernel_launch(void* const* d_in, const int* in_sizes, int n_in,
                              void* d_out, int out_size, void* d_ws, size_t ws_size,
                              hipStream_t stream) {
    const float* x  = (const float*)d_in[0];
    const float* Wk = (const float*)d_in[1];
    const float* Wq = (const float*)d_in[2];
    const float* Wv = (const float*)d_in[3];
    float* out = (float*)d_out;

    const size_t nqkv = (size_t)4 * T_SEQ * D_HEAD;      // 1,048,576 per tensor
    const size_t nx   = (size_t)4 * T_SEQ * E_DIM;       // 8,388,608
    __bf16* kB  = (__bf16*)d_ws;                         // 2 MB
    __bf16* qB  = kB + nqkv;                             // 2 MB
    __bf16* vT  = qB + nqkv;                             // 2 MB
    __bf16* xbf = vT + nqkv;                             // 16 MB
    __bf16* WT  = xbf + nx;                              // 192 KB

    swa_convert_x<<<dim3(nx / (128 * 8)), 128, 0, stream>>>(x, xbf);
    swa_transpose_w<<<dim3(128, 1, 3), 256, 0, stream>>>(Wk, Wq, Wv, WT);

    dim3 g1((4 * T_SEQ) / 16, 1, 3);                     // 1024 M-tiles x {K,Q,V}
    swa_qkv_proj<<<g1, 128, 0, stream>>>(xbf, WT, kB, qB, vT);

    swa_attention<<<256, 128, 0, stream>>>(qB, kB, vT, out);
}
